// EGET_52561809768733
// MI455X (gfx1250) — matrix operand order repaired and checked by reading, not compiled, not run on hardware
//
#include <hip/hip_runtime.h>
#include <hip/hip_bf16.h>
#include <math.h>
#include <stdint.h>

#define N_NODES 20000
#define N_EDGES 400000
#define DIN     1546
#define DINP    1552   // DIN padded to multiple of 8 -> 16B-aligned bf16 rows
#define DH      128
#define DFF     512
#define NLAYERS 2

typedef unsigned short u16;
typedef __attribute__((ext_vector_type(16))) __bf16 v16bf;
typedef __attribute__((ext_vector_type(8)))  float  v8f;

// ---------- bf16 helpers (round-to-nearest-even on store) ----------
__device__ __forceinline__ u16 f2bf_bits(float f) {
  unsigned u = __float_as_uint(f);
  u += 0x7fffu + ((u >> 16) & 1u);
  return (u16)(u >> 16);
}
__device__ __forceinline__ float bf_bits2f(u16 h) {
  return __uint_as_float(((unsigned)h) << 16);
}
__device__ __forceinline__ __bf16 bits2bf(u16 h) {
  __bf16 r; __builtin_memcpy(&r, &h, 2); return r;
}
// monotone float<->uint encoding for atomicMax on floats
__device__ __forceinline__ unsigned encF(float f) {
  int s = __float_as_int(f);
  return (s >= 0) ? ((unsigned)s | 0x80000000u) : ~(unsigned)s;
}
__device__ __forceinline__ float decF(unsigned u) {
  return (u & 0x80000000u) ? __int_as_float((int)(u & 0x7fffffffu))
                           : __int_as_float((int)~u);
}
__device__ __forceinline__ float waveRedSum(float v) {
  #pragma unroll
  for (int o = 16; o > 0; o >>= 1) v += __shfl_xor(v, o, 32);
  return v;
}
__device__ __forceinline__ float waveRedMax(float v) {
  #pragma unroll
  for (int o = 16; o > 0; o >>= 1) v = fmaxf(v, __shfl_xor(v, o, 32));
  return v;
}

// ---------- elementwise f32 -> bf16 (contiguous) ----------
__global__ __launch_bounds__(256)
void cvt_kernel(const float* __restrict__ in, u16* __restrict__ out, long n) {
  long i = (long)blockIdx.x * 256 + threadIdx.x;
  if (i < n) out[i] = f2bf_bits(in[i]);
}

// ---------- 2D f32 -> bf16 with output-row padding ----------
__global__ __launch_bounds__(256)
void cvt2d_kernel(const float* __restrict__ in, int ldin,
                  u16* __restrict__ out, int ldout, long rows, int cols) {
  long i = (long)blockIdx.x * 256 + threadIdx.x;
  if (i >= rows * (long)cols) return;
  long r = i / cols;
  int  c = (int)(i - r * cols);
  out[r * (long)ldout + c] = f2bf_bits(in[r * (long)ldin + c]);
}

// ---------- pack 6 projection weights into [K=1546][768] bf16 ----------
__global__ __launch_bounds__(256)
void pack_w6_kernel(const float* __restrict__ Wq, const float* __restrict__ Wk,
                    const float* __restrict__ Wv, const float* __restrict__ Wr,
                    const float* __restrict__ Whi, const float* __restrict__ Whj,
                    u16* __restrict__ out) {
  int idx = blockIdx.x * 256 + threadIdx.x;
  if (idx >= DIN * DH) return;
  int k = idx >> 7, n = idx & 127;
  size_t o = (size_t)k * 768 + n;
  out[o + 0]   = f2bf_bits(Wq[idx]);
  out[o + 128] = f2bf_bits(Wk[idx]);
  out[o + 256] = f2bf_bits(Wv[idx]);
  out[o + 384] = f2bf_bits(Wr[idx]);
  out[o + 512] = f2bf_bits(Whi[idx]);
  out[o + 640] = f2bf_bits(Whj[idx]);
}

__global__ __launch_bounds__(256)
void pack_bias6_kernel(const float* __restrict__ bq, const float* __restrict__ bk,
                       const float* __restrict__ bv, const float* __restrict__ br,
                       float* __restrict__ out) {
  int t = blockIdx.x * 256 + threadIdx.x;
  if (t >= 768) return;
  int j = t >> 7, n = t & 127;
  float v = 0.f;
  if      (j == 0) v = bq[n];
  else if (j == 1) v = bk[n];
  else if (j == 2) v = bv[n];
  else if (j == 3) v = br[n];
  out[t] = v;
}

// ---------- tiled bf16 WMMA GEMM: C = act(A[MxK] @ B[KxN] + bias) ----------
// block: 128 threads = 4 waves; tile 64(M) x 64(N), K step 32, double-buffered LDS.
// Fast path stages tiles with CDNA5 async global->LDS b128 (ASYNCcnt), one barrier/step.
// REQUIREMENT: lda % 8 == 0 and ldb % 8 == 0 (all internal buffers padded).
#define BM 64
#define BN 64
#define BK 32
#define APAD 8   // row stride 40 u16 = 80 B  (16B multiple)
#define BPAD 8   // row stride 72 u16 = 144 B (16B multiple)

__device__ __forceinline__ void asyncLds128(unsigned ldsOff, const u16* gp) {
  unsigned long long ga = (unsigned long long)gp;
  asm volatile("global_load_async_to_lds_b128 %0, %1, off"
               :: "v"(ldsOff), "v"(ga) : "memory");
}
__device__ __forceinline__ void waitAsync0() {
  asm volatile("s_wait_asynccnt 0x0" ::: "memory");
}

template<int ACT, int OUTBF, bool FAST>   // ACT: 0 none, 1 relu, 2 leaky(0.01)
__device__ __forceinline__ void gemm_core(
    const u16* __restrict__ A, int lda,
    const u16* __restrict__ B, int ldb,
    const float* __restrict__ bias,
    float* __restrict__ Cf, u16* __restrict__ Cb, int ldc,
    int M, int Ndim, int K, int m0, int n0,
    u16 (&As)[2][BM][BK + APAD], u16 (&Bs)[2][BK][BN + BPAD]) {
  int tid  = threadIdx.x;
  int lane = tid & 31;
  int wave = tid >> 5;
  const int ar0 = tid >> 2, ac = (tid & 3) * 8;  // A: rows ar0, ar0+32; 8 cols
  const int br0 = tid >> 3, bc = (tid & 7) * 8;  // B: rows br0, br0+16; 8 cols

  const int kFull  = K / BK;
  const int kTiles = (K + BK - 1) / BK;

  auto stageAsync = [&](int kt, int buf) {
    int k0 = kt * BK;
    #pragma unroll
    for (int it = 0; it < 2; ++it) {
      int r = ar0 + it * 32;
      asyncLds128((unsigned)(uintptr_t)&As[buf][r][ac],
                  A + (size_t)(m0 + r) * lda + k0 + ac);
    }
    #pragma unroll
    for (int it = 0; it < 2; ++it) {
      int r = br0 + it * 16;
      asyncLds128((unsigned)(uintptr_t)&Bs[buf][r][bc],
                  B + (size_t)(k0 + r) * ldb + n0 + bc);
    }
  };
  auto stageGuard = [&](int kt, int buf) {
    int k0 = kt * BK;
    #pragma unroll
    for (int it = 0; it < 2; ++it) {
      int r = ar0 + it * 32;
      int grc = min(m0 + r, M - 1);
      #pragma unroll
      for (int j = 0; j < 8; ++j) {
        int gk = k0 + ac + j;
        u16 val = A[(size_t)grc * lda + min(gk, K - 1)];
        As[buf][r][ac + j] = (m0 + r < M && gk < K) ? val : (u16)0;
      }
    }
    #pragma unroll
    for (int it = 0; it < 2; ++it) {
      int r = br0 + it * 16;
      int gk = k0 + r;
      int gkc = min(gk, K - 1);
      #pragma unroll
      for (int j = 0; j < 8; ++j) {
        int gn = n0 + bc + j;
        u16 val = B[(size_t)gkc * ldb + min(gn, Ndim - 1)];
        Bs[buf][r][bc + j] = (gk < K && gn < Ndim) ? val : (u16)0;
      }
    }
  };

  v8f acc[4] = {};

  if (FAST && kFull > 0) stageAsync(0, 0); else stageGuard(0, 0);
  if (FAST) waitAsync0();
  __syncthreads();

  for (int kt = 0; kt < kTiles; ++kt) {
    int cur = kt & 1, nxt = cur ^ 1;
    if (kt + 1 < kTiles) {
      if (FAST && (kt + 1) < kFull) stageAsync(kt + 1, nxt);
      else                          stageGuard(kt + 1, nxt);
    }
    // A fragment (ISA 7.12.2: 16-bit A 16x32 layout; each half = 8 contiguous u16)
    int lh = lane >> 4, mr = lane & 15;
    v16bf afrag;
    #pragma unroll
    for (int s = 0; s < 16; ++s) {
      int vg = s >> 1, half = s & 1;
      int kk = (vg < 4 ? 2 * vg + half : 16 + 2 * (vg - 4) + half) + 8 * lh;
      afrag[s] = bits2bf(As[cur][wave * 16 + mr][kk]);
    }
    #pragma unroll
    for (int nt = 0; nt < 4; ++nt) {
      v16bf bfrag;
      #pragma unroll
      for (int s = 0; s < 16; ++s)
        bfrag[s] = bits2bf(Bs[cur][(s < 8 ? s : s + 8) + 8 * lh][nt * 16 + mr]);
      acc[nt] = __builtin_amdgcn_wmma_f32_16x16x32_bf16(
          false, afrag, false, bfrag, (short)0, acc[nt], false, false);
    }
    if (FAST) waitAsync0();
    __syncthreads();
  }

  // epilogue: bias + activation; C/D layout M = r + 8*(lane>=16), N = lane&15
  int lh = lane >> 4, nc = lane & 15;
  #pragma unroll
  for (int nt = 0; nt < 4; ++nt) {
    int gn = n0 + nt * 16 + nc;
    if (gn >= Ndim) continue;
    float bvv = bias ? bias[gn] : 0.f;
    #pragma unroll
    for (int r = 0; r < 8; ++r) {
      int gm = m0 + wave * 16 + r + 8 * lh;
      if (gm < M) {
        float val = acc[nt][r] + bvv;
        if (ACT == 1) val = fmaxf(val, 0.f);
        else if (ACT == 2) val = (val > 0.f) ? val : 0.01f * val;
        if (OUTBF) Cb[(size_t)gm * ldc + gn] = f2bf_bits(val);
        else       Cf[(size_t)gm * ldc + gn] = val;
      }
    }
  }
}

template<int ACT, int OUTBF>
__global__ __launch_bounds__(128)
void gemm_bf16_kernel(const u16* __restrict__ A, int lda,
                      const u16* __restrict__ B, int ldb,
                      const float* __restrict__ bias,
                      float* __restrict__ Cf, u16* __restrict__ Cb, int ldc,
                      int M, int Ndim, int K) {
  __shared__ u16 As[2][BM][BK + APAD];
  __shared__ u16 Bs[2][BK][BN + BPAD];
  int m0 = blockIdx.y * BM;
  int n0 = blockIdx.x * BN;
  if ((m0 + BM <= M) && (n0 + BN <= Ndim))
    gemm_core<ACT, OUTBF, true >(A, lda, B, ldb, bias, Cf, Cb, ldc, M, Ndim, K, m0, n0, As, Bs);
  else
    gemm_core<ACT, OUTBF, false>(A, lda, B, ldb, bias, Cf, Cb, ldc, M, Ndim, K, m0, n0, As, Bs);
}

// ---------- edge kernels ----------
// out6 layout per node: [q(0) | k(128) | v(256) | r->h(384) | hi(512) | hj(640)]
__global__ __launch_bounds__(128)
void edge_gate_logit_kernel(const float* __restrict__ ea,
                            const long long* __restrict__ ei,
                            const float* __restrict__ out6,
                            u16* __restrict__ gateb,
                            float* __restrict__ logits) {
  int e = blockIdx.x, t = threadIdx.x;
  int lane = t & 31, w = t >> 5;
  long long s = ei[e], d = ei[N_EDGES + e];
  size_t so = (size_t)s * 768, dofs = (size_t)d * 768;
  float g = ea[(size_t)e * 128 + t] + out6[so + 512 + t] + out6[dofs + 640 + t];
  g = 1.f / (1.f + __expf(-g));
  gateb[(size_t)e * 128 + t] = f2bf_bits(g);
  float qk = out6[dofs + t] * out6[so + 128 + t];
  qk = waveRedSum(qk);
  __shared__ float wred[4];
  if (lane == 0) wred[w] = qk;
  __syncthreads();
  if (t == 0) logits[e] = wred[0] + wred[1] + wred[2] + wred[3];
}

__global__ void init_stats_kernel(unsigned* stats) { stats[0] = 0u; ((float*)stats)[1] = 0.f; }

__global__ __launch_bounds__(256)
void reduce_max_kernel(const float* __restrict__ l, int n, unsigned* __restrict__ maxenc) {
  float m = -3.4e38f;
  for (int i = blockIdx.x * 256 + threadIdx.x; i < n; i += gridDim.x * 256)
    m = fmaxf(m, l[i]);
  m = waveRedMax(m);
  __shared__ float wred[8];
  int lane = threadIdx.x & 31, w = threadIdx.x >> 5;
  if (lane == 0) wred[w] = m;
  __syncthreads();
  if (threadIdx.x == 0) {
    float mm = wred[0];
    #pragma unroll
    for (int i = 1; i < 8; ++i) mm = fmaxf(mm, wred[i]);
    atomicMax(maxenc, encF(mm));
  }
}

__global__ __launch_bounds__(256)
void expsum_kernel(const float* __restrict__ l, int n,
                   const unsigned* __restrict__ maxenc, float* __restrict__ sum) {
  float mx = decF(*maxenc);
  float a = 0.f;
  for (int i = blockIdx.x * 256 + threadIdx.x; i < n; i += gridDim.x * 256)
    a += __expf(l[i] - mx);
  a = waveRedSum(a);
  __shared__ float wred[8];
  int lane = threadIdx.x & 31, w = threadIdx.x >> 5;
  if (lane == 0) wred[w] = a;
  __syncthreads();
  if (threadIdx.x == 0) {
    float ss = 0.f;
    #pragma unroll
    for (int i = 0; i < 8; ++i) ss += wred[i];
    atomicAdd(sum, ss);
  }
}

__global__ __launch_bounds__(128)
void scatter_kernel(const long long* __restrict__ ei, const float* __restrict__ logits,
                    const unsigned* __restrict__ stats_u, const float* __restrict__ stats_f,
                    const u16* __restrict__ gateb, float* __restrict__ out6) {
  int e = blockIdx.x, t = threadIdx.x;
  float mx  = decF(stats_u[0]);
  float inv = 1.f / stats_f[1];
  float p = __expf(logits[e] - mx) * inv;
  long long s = ei[e], d = ei[N_EDGES + e];
  float m = p * out6[(size_t)s * 768 + 256 + t] * bf_bits2f(gateb[(size_t)e * 128 + t]);
  atomicAdd(&out6[(size_t)d * 768 + 384 + t], m);   // adds onto x@Wr + br slice
}

// ---------- LayerNorm over 128-wide rows; optional residual add; bf16 out ----------
__global__ __launch_bounds__(128)
void ln_kernel(const float* __restrict__ X, int strideX, const float* __restrict__ Add,
               const float* __restrict__ g, const float* __restrict__ b,
               u16* __restrict__ outb) {
  int row = blockIdx.x, t = threadIdx.x;
  int lane = t & 31, w = t >> 5;
  float v = X[(size_t)row * strideX + t];
  if (Add) v += Add[(size_t)row * 128 + t];
  __shared__ float w0[4], w1[4];
  float s = waveRedSum(v);
  if (lane == 0) w0[w] = s;
  __syncthreads();
  float mean = (w0[0] + w0[1] + w0[2] + w0[3]) * (1.f / 128.f);
  float dv = v - mean;
  s = waveRedSum(dv * dv);
  if (lane == 0) w1[w] = s;
  __syncthreads();
  float rstd = rsqrtf((w1[0] + w1[1] + w1[2] + w1[3]) * (1.f / 128.f) + 1e-5f);
  outb[(size_t)row * 128 + t] = f2bf_bits(dv * rstd * g[t] + b[t]);
}

// ---------- host ----------
extern "C" void kernel_launch(void* const* d_in, const int* in_sizes, int n_in,
                              void* d_out, int out_size, void* d_ws, size_t ws_size,
                              hipStream_t stream) {
  (void)in_sizes; (void)n_in; (void)out_size; (void)ws_size;
  const float*     x     = (const float*)d_in[0];
  const long long* ei    = (const long long*)d_in[1];   // int64 edge_index [2,E]
  const float*     ea    = (const float*)d_in[2];
  const float* Wq  = (const float*)d_in[3];  const float* bq  = (const float*)d_in[4];
  const float* Wk  = (const float*)d_in[5];  const float* bk  = (const float*)d_in[6];
  const float* Wv  = (const float*)d_in[7];  const float* bv  = (const float*)d_in[8];
  const float* Wr  = (const float*)d_in[9];  const float* br  = (const float*)d_in[10];
  const float* Whi = (const float*)d_in[11]; const float* Whj = (const float*)d_in[12];
  const float* g1  = (const float*)d_in[13]; const float* be1 = (const float*)d_in[14];
  const float* W1  = (const float*)d_in[15]; const float* bl1 = (const float*)d_in[16];
  const float* W2  = (const float*)d_in[17]; const float* bl2 = (const float*)d_in[18];
  const float* g2  = (const float*)d_in[19]; const float* be2 = (const float*)d_in[20];
  const float* Wlin  = (const float*)d_in[21]; const float* blin  = (const float*)d_in[22];
  const float* Wlin2 = (const float*)d_in[23]; const float* blin2 = (const float*)d_in[24];

  char* ws = (char*)d_ws;
  size_t off = 0;
  auto alloc = [&](size_t bytes) -> char* {
    char* p = ws + off;
    off += (bytes + 255) & ~(size_t)255;
    return p;
  };
  u16*   xbA    = (u16*)  alloc((size_t)N_NODES * DINP * 2);
  u16*   xbB    = (u16*)  alloc((size_t)N_NODES * DINP * 2);
  u16*   w6b    = (u16*)  alloc((size_t)DIN * 768 * 2);
  float* bias6  = (float*)alloc(768 * 4);
  float* out6   = (float*)alloc((size_t)N_NODES * 768 * 4);
  u16*   gateb  = (u16*)  alloc((size_t)N_EDGES * 128 * 2);
  float* logits = (float*)alloc((size_t)N_EDGES * 4);
  unsigned* stats = (unsigned*)alloc(256);
  u16*   ssb    = (u16*)  alloc((size_t)N_NODES * 128 * 2);
  u16*   ffb    = (u16*)  alloc((size_t)N_NODES * 512 * 2);
  float* ss2    = (float*)alloc((size_t)N_NODES * 128 * 4);
  u16*   hb     = (u16*)  alloc((size_t)N_NODES * 128 * 2);
  u16*   W1b    = (u16*)  alloc((size_t)128 * 512 * 2);
  u16*   W2b    = (u16*)  alloc((size_t)512 * 128 * 2);
  u16*   Wlinb  = (u16*)  alloc((size_t)128 * DINP * 2);
  u16*   Wlin2b = (u16*)  alloc((size_t)DIN * 128 * 2);

  auto cdiv = [](long a, long b) { return (int)((a + b - 1) / b); };

  // one-time conversions (row-padded where the padded dim is a GEMM leading dim)
  cvt2d_kernel<<<cdiv((long)N_NODES * DIN, 256), 256, 0, stream>>>(x, DIN, xbA, DINP, N_NODES, DIN);
  cvt2d_kernel<<<cdiv((long)128 * DIN, 256), 256, 0, stream>>>(Wlin, DIN, Wlinb, DINP, 128, DIN);
  cvt_kernel<<<cdiv((long)DIN * 128, 256), 256, 0, stream>>>(Wlin2, Wlin2b, (long)DIN * 128);

  u16* xcur = xbA;
  u16* xnext = xbB;
  for (int i = 0; i < NLAYERS; ++i) {
    size_t wOff = (size_t)i * DIN * DH;
    pack_w6_kernel<<<cdiv((long)DIN * DH, 256), 256, 0, stream>>>(
        Wq + wOff, Wk + wOff, Wv + wOff, Wr + wOff, Whi + wOff, Whj + wOff, w6b);
    pack_bias6_kernel<<<3, 256, 0, stream>>>(bq + (size_t)i * 128, bk + (size_t)i * 128,
                                             bv + (size_t)i * 128, br + (size_t)i * 128, bias6);
    cvt_kernel<<<cdiv((long)128 * 512, 256), 256, 0, stream>>>(W1 + (size_t)i * 128 * 512, W1b, 128 * 512);
    cvt_kernel<<<cdiv((long)512 * 128, 256), 256, 0, stream>>>(W2 + (size_t)i * 512 * 128, W2b, 512 * 128);

    // fused q|k|v|r|hi|hj projections: [N,1546] @ [1546,768]
    {
      dim3 g(cdiv(768, BN), cdiv(N_NODES, BM));
      gemm_bf16_kernel<0, 0><<<g, 128, 0, stream>>>(xcur, DINP, w6b, 768, bias6,
                                                    out6, nullptr, 768, N_NODES, 768, DIN);
    }
    // attention: gate + logits, global softmax stats, weighted scatter-add onto r slice
    edge_gate_logit_kernel<<<N_EDGES, 128, 0, stream>>>(ea, ei, out6, gateb, logits);
    init_stats_kernel<<<1, 1, 0, stream>>>(stats);
    reduce_max_kernel<<<1024, 256, 0, stream>>>(logits, N_EDGES, stats);
    expsum_kernel<<<1024, 256, 0, stream>>>(logits, N_EDGES, stats, (float*)stats + 1);
    scatter_kernel<<<N_EDGES, 128, 0, stream>>>(ei, logits, stats, (const float*)stats, gateb, out6);

    // FFMLP: pre-norm residual
    ln_kernel<<<N_NODES, 128, 0, stream>>>(out6 + 384, 768, nullptr,
                                           g1 + (size_t)i * 128, be1 + (size_t)i * 128, ssb);
    {
      dim3 g(cdiv(DFF, BN), cdiv(N_NODES, BM));
      gemm_bf16_kernel<1, 1><<<g, 128, 0, stream>>>(ssb, 128, W1b, DFF, bl1 + (size_t)i * DFF,
                                                    nullptr, ffb, DFF, N_NODES, DFF, 128);
    }
    {
      dim3 g(cdiv(DH, BN), cdiv(N_NODES, BM));
      gemm_bf16_kernel<0, 0><<<g, 128, 0, stream>>>(ffb, DFF, W2b, DH, bl2 + (size_t)i * 128,
                                                    ss2, nullptr, DH, N_NODES, DH, DFF);
    }
    ln_kernel<<<N_NODES, 128, 0, stream>>>(out6 + 384, 768, ss2,
                                           g2 + (size_t)i * 128, be2 + (size_t)i * 128, hb);
    // expand back to 1546 (bf16, padded rows, for next layer's GEMMs)
    {
      dim3 g(cdiv(DIN, BN), cdiv(N_NODES, BM));
      gemm_bf16_kernel<0, 1><<<g, 128, 0, stream>>>(hb, 128, Wlinb, DINP, blin,
                                                    nullptr, xnext, DINP, N_NODES, DIN, 128);
    }
    u16* tmp = xcur; xcur = xnext; xnext = tmp;
  }
  // final head: leaky_relu(x @ Wlin2 + blin2) -> fp32 output [N,128]
  {
    dim3 g(cdiv(DH, BN), cdiv(N_NODES, BM));
    gemm_bf16_kernel<2, 0><<<g, 128, 0, stream>>>(xcur, DINP, Wlin2b, DH, blin2,
                                                  (float*)d_out, nullptr, DH, N_NODES, DH, DIN);
  }
}
